// RelGraphConv1_37271726195521
// MI455X (gfx1250) — compile-verified, matrix operand order checked by reading, not run on hardware
//
#include <hip/hip_runtime.h>
#include <hip/hip_bf16.h>
#include <stdint.h>

typedef __attribute__((ext_vector_type(16))) __bf16 v16bf;
typedef __attribute__((ext_vector_type(8)))  float  v8f;

#define IN_FEAT   128
#define OUT_FEAT  128
#define NUM_RELS  16
#define NUM_BASES 8

__device__ __forceinline__ unsigned short f2bf(float f) {
  unsigned int u = __float_as_uint(f);
  u += 0x7FFFu + ((u >> 16) & 1u);      // round-to-nearest-even
  return (unsigned short)(u >> 16);
}

// ---------------- control / sort-by-relation kernels ----------------
// ctrl[0..15]  = histogram
// ctrl[16..32] = padded segment offsets (offs[16] = padded total)
// ctrl[64..79] = scatter cursors
__global__ void k_zero_ctrl(int* ctrl) {
  int i = threadIdx.x;
  if (i < 128) ctrl[i] = 0;
}

__global__ void k_hist(const int* __restrict__ et, int E, int* __restrict__ hist) {
  int e = blockIdx.x * blockDim.x + threadIdx.x;
  if (e < E) atomicAdd(&hist[et[e]], 1);
}

__global__ void k_scan(int* ctrl) {
  int* hist = ctrl;
  int* offs = ctrl + 16;
  int* cur  = ctrl + 64;
  int acc = 0;
  for (int r = 0; r < NUM_RELS; ++r) {
    offs[r] = acc;
    cur[r]  = acc;
    acc += (hist[r] + 15) & ~15;        // pad each segment to 16 edges
  }
  offs[NUM_RELS] = acc;
}

__global__ void k_fill(int* __restrict__ psrc, int* __restrict__ pdst, int n) {
  int i = blockIdx.x * blockDim.x + threadIdx.x;
  if (i < n) { psrc[i] = 0; pdst[i] = -1; }   // -1 = padding sentinel
}

__global__ void k_scatter(const int* __restrict__ src, const int* __restrict__ dst,
                          const int* __restrict__ et, int E, int* __restrict__ cur,
                          int* __restrict__ psrc, int* __restrict__ pdst) {
  int e = blockIdx.x * blockDim.x + threadIdx.x;
  if (e < E) {
    int r = et[e];
    int p = atomicAdd(&cur[r], 1);
    psrc[p] = src[e];
    pdst[p] = dst[e];
  }
}

// ---------------- weight combine + pack into WMMA B layout ----------------
// Wpack[r][kt(4)][nt(8)][lane(32)][j(16)] : lane<16 -> N=lane, K pairs 0-7,16-23 (+32*kt)
//                                           lane>=16 -> N=lane-16, K pairs 8-15,24-31
__global__ void k_wpack(const float* __restrict__ weight, const float* __restrict__ w_comp,
                        unsigned short* __restrict__ Wpack) {
  int idx = blockIdx.x * blockDim.x + threadIdx.x;
  if (idx >= NUM_RELS * 4 * 8 * 32 * 16) return;
  int j    = idx & 15;
  int lane = (idx >> 4) & 31;
  int nt   = (idx >> 9) & 7;
  int kt   = (idx >> 12) & 3;
  int r    = (idx >> 14);
  int hi   = lane >> 4;
  int K = kt * 32 + 8 * hi + ((j < 8) ? j : (j + 8));
  int N = nt * 16 + (lane & 15);
  float acc = 0.f;
  #pragma unroll
  for (int b = 0; b < NUM_BASES; ++b)
    acc += w_comp[r * NUM_BASES + b] * weight[((size_t)b * IN_FEAT + K) * OUT_FEAT + N];
  Wpack[idx] = f2bf(acc);
}

__global__ void k_xbf16(const float* __restrict__ x, unsigned short* __restrict__ xb, int n) {
  int i = blockIdx.x * blockDim.x + threadIdx.x;
  if (i < n) xb[i] = f2bf(x[i]);
}

__global__ void k_init_out(float* __restrict__ out, const float* __restrict__ bias, int n) {
  int i = blockIdx.x * blockDim.x + threadIdx.x;
  if (i < n) out[i] = bias[i & (OUT_FEAT - 1)];
}

// ---------------- main: one wave per 16-edge tile, 32x WMMA bf16 ----------------
__global__ void __launch_bounds__(256)
k_rgcn_wmma(const unsigned short* __restrict__ xb,
            const unsigned short* __restrict__ Wpack,
            const int* __restrict__ psrc, const int* __restrict__ pdst,
            const int* __restrict__ ctrl, float* __restrict__ out) {
  const int lane = threadIdx.x & 31;
  const int tile = (int)((blockIdx.x * blockDim.x + threadIdx.x) >> 5);
  const int total = ctrl[16 + NUM_RELS];
  const int base  = tile * 16;
  if (base >= total) return;            // wave-uniform: EXEC stays all-ones

  // which relation owns this (16-aligned) tile
  const int* offs = ctrl + 16;
  int r = 0;
  #pragma unroll
  for (int i = 1; i < NUM_RELS; ++i) r += (base >= offs[i]) ? 1 : 0;

  const int m  = lane & 15;
  const int hi = lane >> 4;
  const int srow = psrc[base + m];
  const int drow = pdst[base + m];

  union V { unsigned int u[8]; uint4 q[2]; v16bf v; };

  // Gather A: 16 rows x 128 feats of bf16 x, in WMMA A layout, 4 K-slices of 32
  v16bf A[4];
  const unsigned short* xrow = xb + (size_t)srow * IN_FEAT;
  #pragma unroll
  for (int kt = 0; kt < 4; ++kt) {
    V a;
    #pragma unroll
    for (int jj = 0; jj < 8; ++jj) {
      int j  = jj * 2;
      int k0 = kt * 32 + 8 * hi + ((j < 8) ? j : (j + 8));
      a.u[jj] = *(const unsigned int*)(xrow + k0);   // 2 contiguous bf16
    }
    A[kt] = a.v;
  }

  v8f C[8];
  #pragma unroll
  for (int nt = 0; nt < 8; ++nt) { v8f z = {}; C[nt] = z; }

  #pragma unroll
  for (int nt = 0; nt < 8; ++nt) {
    #pragma unroll
    for (int kt = 0; kt < 4; ++kt) {
      V bm;
      const uint4* bp = (const uint4*)(Wpack +
          ((((size_t)r * 4 + kt) * 8 + nt) * 32 + lane) * 16);
      bm.q[0] = bp[0];
      bm.q[1] = bp[1];
      C[nt] = __builtin_amdgcn_wmma_f32_16x16x32_bf16(
                  false, A[kt], false, bm.v, (short)0, C[nt], false, false);
    }
  }

  // Scatter-add: lane holds column nt*16+m of rows (hi*8 .. hi*8+7)
  #pragma unroll
  for (int nt = 0; nt < 8; ++nt) {
    int col = nt * 16 + m;
    #pragma unroll
    for (int rr = 0; rr < 8; ++rr) {
      int mrow = rr + hi * 8;
      int d = __shfl(drow, mrow, 32);
      if (d >= 0)
        unsafeAtomicAdd(&out[(size_t)d * OUT_FEAT + col], C[nt][rr]);
    }
  }
}

// ---------------- host ----------------
extern "C" void kernel_launch(void* const* d_in, const int* in_sizes, int n_in,
                              void* d_out, int out_size, void* d_ws, size_t ws_size,
                              hipStream_t stream) {
  const float* x      = (const float*)d_in[0];
  const float* weight = (const float*)d_in[1];
  const float* w_comp = (const float*)d_in[2];
  const float* h_bias = (const float*)d_in[3];
  const int*   src    = (const int*)d_in[4];
  const int*   dst    = (const int*)d_in[5];
  const int*   et     = (const int*)d_in[6];
  const int E  = in_sizes[4];
  const int NX = in_sizes[0];           // N_NODES * IN_FEAT
  float* out = (float*)d_out;

  // workspace layout (256B aligned slabs)
  size_t off = 0;
  auto a256 = [](size_t v) { return (v + 255) & ~(size_t)255; };
  int* ctrl = (int*)((char*)d_ws + off);
  off = a256(off + 128 * sizeof(int));
  unsigned short* Wpack = (unsigned short*)((char*)d_ws + off);
  off = a256(off + (size_t)NUM_RELS * 4 * 8 * 32 * 16 * 2);
  unsigned short* xb = (unsigned short*)((char*)d_ws + off);
  off = a256(off + (size_t)NX * 2);
  const int padded = E + 16 * NUM_RELS;
  int* psrc = (int*)((char*)d_ws + off);
  off = a256(off + (size_t)padded * 4);
  int* pdst = (int*)((char*)d_ws + off);
  off = a256(off + (size_t)padded * 4);
  (void)ws_size; (void)n_in;

  k_zero_ctrl<<<1, 128, 0, stream>>>(ctrl);
  k_hist<<<(E + 255) / 256, 256, 0, stream>>>(et, E, ctrl);
  k_scan<<<1, 1, 0, stream>>>(ctrl);
  k_fill<<<(padded + 255) / 256, 256, 0, stream>>>(psrc, pdst, padded);
  k_scatter<<<(E + 255) / 256, 256, 0, stream>>>(src, dst, et, E, ctrl + 64, psrc, pdst);
  k_wpack<<<(NUM_RELS * 4 * 8 * 32 * 16 + 255) / 256, 256, 0, stream>>>(weight, w_comp, Wpack);
  k_xbf16<<<(NX + 255) / 256, 256, 0, stream>>>(x, xb, NX);
  k_init_out<<<(out_size + 255) / 256, 256, 0, stream>>>(out, h_bias, out_size);

  const int tiles   = (padded + 15) / 16;
  const int threads = tiles * 32;       // one wave32 per 16-edge tile
  k_rgcn_wmma<<<(threads + 255) / 256, 256, 0, stream>>>(xb, Wpack, psrc, pdst, ctrl, out);
}
